// QuantVGG11_66331474920078
// MI455X (gfx1250) — 8-bit matrix operand order repaired and checked by reading, compiled, not run on hardware
//
#include <hip/hip_runtime.h>
#include <math.h>


typedef int v8i __attribute__((ext_vector_type(8)));
union FragI { v8i v; int e[8]; };

#define TPB 256


__global__ void zero_u32_kernel(unsigned* p) { *p = 0u; }

__global__ void finalize_scale_kernel(const unsigned* bits, float n, float* scale) {
    *scale = fmaxf(__uint_as_float(*bits), 1e-8f) / n;
}

__global__ void absmax_kernel(const float* __restrict__ x, int n, unsigned* bits) {
    __shared__ unsigned sm[TPB];
    float m = 0.f;
    for (int i = blockIdx.x * blockDim.x + threadIdx.x; i < n; i += gridDim.x * blockDim.x)
        m = fmaxf(m, fabsf(x[i]));
    sm[threadIdx.x] = __float_as_uint(m);
    __syncthreads();
    for (int off = TPB / 2; off > 0; off >>= 1) {
        if (threadIdx.x < off)
            sm[threadIdx.x] = sm[threadIdx.x] > sm[threadIdx.x + off] ? sm[threadIdx.x]
                                                                      : sm[threadIdx.x + off];
        __syncthreads();
    }
    if (threadIdx.x == 0) atomicMax(bits, sm[0]);
}

__global__ void relumax_kernel(const float* __restrict__ x, int n, unsigned* bits) {
    __shared__ unsigned sm[TPB];
    float m = 0.f;
    for (int i = blockIdx.x * blockDim.x + threadIdx.x; i < n; i += gridDim.x * blockDim.x)
        m = fmaxf(m, x[i]);
    sm[threadIdx.x] = __float_as_uint(m);
    __syncthreads();
    for (int off = TPB / 2; off > 0; off >>= 1) {
        if (threadIdx.x < off)
            sm[threadIdx.x] = sm[threadIdx.x] > sm[threadIdx.x + off] ? sm[threadIdx.x]
                                                                      : sm[threadIdx.x + off];
        __syncthreads();
    }
    if (threadIdx.x == 0) atomicMax(bits, sm[0]);
}


__global__ void quant_input_kernel(const float* __restrict__ x, const float* xs_p,
                                   signed char* __restrict__ q,
                                   int B, int C, int H, int W) {
    int total = B * C * H * W;
    int i = blockIdx.x * blockDim.x + threadIdx.x;
    if (i >= total) return;
    int c = i % C;
    int w = (i / C) % W;
    int h = (i / (C * W)) % H;
    int b = i / (C * W * H);
    float v = x[((b * C + c) * H + h) * W + w];
    float r = rintf(v / xs_p[0]);
    r = fminf(fmaxf(r, -8.f), 7.f);
    q[i] = (signed char)r;
}

__global__ void quant_conv_w_kernel(const float* __restrict__ w, const float* ws_p,
                                    signed char* __restrict__ wq, int Co, int Ci) {
    int K = 9 * Ci;
    int total = Co * K;
    int i = blockIdx.x * blockDim.x + threadIdx.x;
    if (i >= total) return;
    int o = i / K;
    int rem = i % K;
    int ci = rem % Ci;
    int t = rem / Ci;
    int kw = t % 3, kh = t / 3;
    float v = w[((o * Ci + ci) * 3 + kh) * 3 + kw];
    float r = rintf(v / ws_p[0]);
    r = fminf(fmaxf(r, -7.f), 7.f);
    wq[i] = (signed char)r;
}

__global__ void quant_lin_w_kernel(const float* __restrict__ w, const float* ws_p,
                                   signed char* __restrict__ wq, int total) {
    int i = blockIdx.x * blockDim.x + threadIdx.x;
    if (i >= total) return;
    float r = rintf(w[i] / ws_p[0]);
    r = fminf(fmaxf(r, -7.f), 7.f);
    wq[i] = (signed char)r;
}

__global__ void bias_quant_kernel(const float* __restrict__ b, const float* xs_p,
                                  const float* ws_p, float* __restrict__ bf, int n) {
    int i = blockIdx.x * blockDim.x + threadIdx.x;
    if (i >= n) return;
    float s = xs_p[0] * ws_p[0];
    bf[i] = rintf(b[i] / s) * s;
}

__global__ void relu_quant_kernel(const float* __restrict__ y, const float* xs_p,
                                  signed char* __restrict__ q, int n) {
    int i = blockIdx.x * blockDim.x + threadIdx.x;
    if (i >= n) return;
    float r = rintf(fmaxf(y[i], 0.f) / xs_p[0]);
    r = fminf(fmaxf(r, 0.f), 15.f);
    q[i] = (signed char)r;
}

__global__ void maxpool_kernel(const signed char* __restrict__ in,
                               signed char* __restrict__ out,
                               int B, int H, int W, int C) {
    int Ho = H >> 1, Wo = W >> 1;
    int total = B * Ho * Wo * C;
    int i = blockIdx.x * blockDim.x + threadIdx.x;
    if (i >= total) return;
    int c = i % C;
    int wo = (i / C) % Wo;
    int ho = (i / (C * Wo)) % Ho;
    int b = i / (C * Wo * Ho);
    const signed char* p = in + (((b * H + 2 * ho) * W) + 2 * wo) * C + c;
    int rowStride = W * C;
    signed char m = p[0];
    signed char v1 = p[C];             m = v1 > m ? v1 : m;
    signed char v2 = p[rowStride];     m = v2 > m ? v2 : m;
    signed char v3 = p[rowStride + C]; m = v3 > m ? v3 : m;
    out[i] = m;
}


__device__ inline void load_a_frag(FragI& a, const int* As, int row, int half) {
#pragma unroll
    for (int v = 0; v < 8; ++v) {
        int g = v & 3;
        int k = ((v >> 2) << 5) + (half << 4) + (g << 2);
        a.e[v] = As[row * 16 + (k >> 2)];
    }
}

__device__ inline void load_b_frag(FragI& b, const int* Bs, int col, int half) {
#pragma unroll
    for (int v = 0; v < 8; ++v) {
        int k = ((v >> 2) << 5) + (half << 4) + ((v & 3) << 2);
        b.e[v] = Bs[col * 16 + (k >> 2)];
    }
}

template <bool SGN_A>
__global__ __launch_bounds__(TPB) void conv_iu8_kernel(
    const signed char* __restrict__ act, const signed char* __restrict__ wq,
    const float* __restrict__ bf, const float* __restrict__ xs_p,
    const float* __restrict__ ws_p, float* __restrict__ y,
    int H, int W, int Ci, int Co) {
    __shared__ alignas(16) int As[128 * 16];
    __shared__ alignas(16) int Bs[64 * 16];

    const int K = 9 * Ci;
    const int M0 = blockIdx.x * 128;
    const int N0 = blockIdx.y * 64;
    const int tid = threadIdx.x;
    const int lane = tid & 31;
    const int wave = tid >> 5;
    const int half = lane >> 4;
    const int l16 = lane & 15;
    const int HW = H * W;
    const bool vecc = (Ci & 63) == 0;

    const int ar = tid >> 1;
    const int aq = (tid & 1) << 5;
    const int am = M0 + ar;
    const int apx = am % W;
    const int apy = (am / W) % H;
    const int aimg = am / HW;
    const int bn = tid >> 2;
    const int bq = (tid & 3) << 4;

    v8i z = {0, 0, 0, 0, 0, 0, 0, 0};
    v8i c0 = z, c1 = z, c2 = z, c3 = z;

    const int nChunks = (K + 63) >> 6;
    for (int kc = 0; kc < nChunks; ++kc) {
        const int kbase = kc << 6;
        __syncthreads();
        {
            signed char* dst = (signed char*)As + ar * 64 + aq;
            if (vecc) {
                int t = kbase / Ci;
                int cib = kbase - t * Ci;
                int kw = t % 3, kh = t / 3;
                int ih = apy + kh - 1, iw = apx + kw - 1;
                if (ih >= 0 && ih < H && iw >= 0 && iw < W) {
                    const signed char* src =
                        act + ((size_t)((aimg * H + ih) * W + iw)) * Ci + cib + aq;
                    ((int4*)dst)[0] = ((const int4*)src)[0];
                    ((int4*)dst)[1] = ((const int4*)src)[1];
                } else {
                    ((int4*)dst)[0] = make_int4(0, 0, 0, 0);
                    ((int4*)dst)[1] = make_int4(0, 0, 0, 0);
                }
            } else {
                for (int bb = 0; bb < 32; ++bb) {
                    int k = kbase + aq + bb;
                    signed char val = 0;
                    if (k < K) {
                        int ci = k % Ci;
                        int t = k / Ci;
                        int kw = t % 3, kh = t / 3;
                        int ih = apy + kh - 1, iw = apx + kw - 1;
                        if (ih >= 0 && ih < H && iw >= 0 && iw < W)
                            val = act[((aimg * H + ih) * W + iw) * Ci + ci];
                    }
                    dst[bb] = val;
                }
            }
        }
        {
            int co = N0 + bn;
            signed char* dst = (signed char*)Bs + bn * 64 + bq;
            const signed char* src = wq + (size_t)co * K + kbase + bq;
            if (((K & 15) == 0) && (kbase + bq + 16 <= K)) {
                *(int4*)dst = *(const int4*)src;
                if (kc + 1 < nChunks) __builtin_prefetch(src + 64, 0, 1);
            } else {
                for (int bb = 0; bb < 16; ++bb) {
                    int k = kbase + bq + bb;
                    dst[bb] = (k < K) ? src[bb] : (signed char)0;
                }
            }
        }
        __syncthreads();

        FragI a;
        load_a_frag(a, As, wave * 16 + l16, half);
        FragI b;
        load_b_frag(b, Bs, 0 + l16, half);
        c0 = __builtin_amdgcn_wmma_i32_16x16x64_iu8(SGN_A, a.v, true, b.v, c0, false, false);
        load_b_frag(b, Bs, 16 + l16, half);
        c1 = __builtin_amdgcn_wmma_i32_16x16x64_iu8(SGN_A, a.v, true, b.v, c1, false, false);
        load_b_frag(b, Bs, 32 + l16, half);
        c2 = __builtin_amdgcn_wmma_i32_16x16x64_iu8(SGN_A, a.v, true, b.v, c2, false, false);
        load_b_frag(b, Bs, 48 + l16, half);
        c3 = __builtin_amdgcn_wmma_i32_16x16x64_iu8(SGN_A, a.v, true, b.v, c3, false, false);
    }

    const float sc = xs_p[0] * ws_p[0];
    FragI r[4];
    r[0].v = c0; r[1].v = c1; r[2].v = c2; r[3].v = c3;
#pragma unroll
    for (int j = 0; j < 4; ++j) {
        int n = N0 + j * 16 + l16;
        float bias = bf[n];
#pragma unroll
        for (int v = 0; v < 8; ++v) {
            int m = M0 + wave * 16 + half * 8 + v;
            y[(size_t)m * Co + n] = (float)r[j].e[v] * sc + bias;
        }
    }
}

__global__ __launch_bounds__(TPB) void gemm_iu8_kernel(
    const signed char* __restrict__ act, const signed char* __restrict__ wq,
    const float* __restrict__ bf, const float* __restrict__ xs_p,
    const float* __restrict__ ws_p, float* __restrict__ y,
    int M, int N, int K) {
    __shared__ alignas(16) int As[128 * 16];
    __shared__ alignas(16) int Bs[64 * 16];

    const int M0 = blockIdx.x * 128;
    const int N0 = blockIdx.y * 64;
    const int tid = threadIdx.x;
    const int lane = tid & 31;
    const int wave = tid >> 5;
    const int half = lane >> 4;
    const int l16 = lane & 15;

    const int ar = tid >> 1;
    const int aq = (tid & 1) << 5;
    const int bn = tid >> 2;
    const int bq = (tid & 3) << 4;

    v8i z = {0, 0, 0, 0, 0, 0, 0, 0};
    v8i c0 = z, c1 = z, c2 = z, c3 = z;

    const int nChunks = K >> 6;
    for (int kc = 0; kc < nChunks; ++kc) {
        const int kbase = kc << 6;
        __syncthreads();
        {
            const signed char* src = act + (size_t)(M0 + ar) * K + kbase + aq;
            signed char* dst = (signed char*)As + ar * 64 + aq;
            ((int4*)dst)[0] = ((const int4*)src)[0];
            ((int4*)dst)[1] = ((const int4*)src)[1];
        }
        {
            int co = N0 + bn;
            signed char* dst = (signed char*)Bs + bn * 64 + bq;
            if (co < N) {
                const signed char* src = wq + (size_t)co * K + kbase + bq;
                *(int4*)dst = *(const int4*)src;
                if (kc + 1 < nChunks) __builtin_prefetch(src + 64, 0, 1);
            } else {
                *(int4*)dst = make_int4(0, 0, 0, 0);
            }
        }
        __syncthreads();

        FragI a;
        load_a_frag(a, As, wave * 16 + l16, half);
        FragI b;
        load_b_frag(b, Bs, 0 + l16, half);
        c0 = __builtin_amdgcn_wmma_i32_16x16x64_iu8(false, a.v, true, b.v, c0, false, false);
        load_b_frag(b, Bs, 16 + l16, half);
        c1 = __builtin_amdgcn_wmma_i32_16x16x64_iu8(false, a.v, true, b.v, c1, false, false);
        load_b_frag(b, Bs, 32 + l16, half);
        c2 = __builtin_amdgcn_wmma_i32_16x16x64_iu8(false, a.v, true, b.v, c2, false, false);
        load_b_frag(b, Bs, 48 + l16, half);
        c3 = __builtin_amdgcn_wmma_i32_16x16x64_iu8(false, a.v, true, b.v, c3, false, false);
    }

    const float sc = xs_p[0] * ws_p[0];
    FragI r[4];
    r[0].v = c0; r[1].v = c1; r[2].v = c2; r[3].v = c3;
#pragma unroll
    for (int j = 0; j < 4; ++j) {
        int n = N0 + j * 16 + l16;
        if (n < N) {
            float bias = bf[n];
#pragma unroll
            for (int v = 0; v < 8; ++v) {
                int m = M0 + wave * 16 + half * 8 + v;
                y[(size_t)m * N + n] = (float)r[j].e[v] * sc + bias;
            }
        }
    }
}

__global__ void log_softmax10_kernel(const float* __restrict__ y, float* __restrict__ out) {
    int m = blockIdx.x * blockDim.x + threadIdx.x;
    if (m >= 256) return;
    const float* r = y + m * 10;
    float mx = r[0];
    for (int j = 1; j < 10; ++j) mx = fmaxf(mx, r[j]);
    float s = 0.f;
    for (int j = 0; j < 10; ++j) s += expf(r[j] - mx);
    float l = logf(s);
    for (int j = 0; j < 10; ++j) out[m * 10 + j] = r[j] - mx - l;
}


static inline int cdiv(int a, int b) { return (a + b - 1) / b; }

extern "C" void kernel_launch(void* const* d_in, const int* in_sizes, int n_in,
                              void* d_out, int out_size, void* d_ws, size_t ws_size,
                              hipStream_t stream) {
    (void)in_sizes; (void)n_in; (void)out_size; (void)ws_size;

    char* p = (char*)d_ws;
    unsigned* bits = (unsigned*)p;
    float* XS = (float*)(p + 4);
    float* WS = (float*)(p + 8);
    char* Y = p + 256;
    const size_t Y_BYTES = 72ull << 20;
    char* A0 = Y + Y_BYTES;
    const size_t A_BYTES = 20ull << 20;
    char* A1 = A0 + A_BYTES;
    char* WQ = A1 + A_BYTES;
    const size_t WQ_BYTES = 20ull << 20;
    float* BF = (float*)(WQ + WQ_BYTES);

    const int B = 256;
    const float* x = (const float*)d_in[0];

    {
        int n = B * 3 * 32 * 32;
        zero_u32_kernel<<<1, 1, 0, stream>>>(bits);
        absmax_kernel<<<512, TPB, 0, stream>>>(x, n, bits);
        finalize_scale_kernel<<<1, 1, 0, stream>>>(bits, 7.0f, XS);
        quant_input_kernel<<<cdiv(n, TPB), TPB, 0, stream>>>(
            x, XS, (signed char*)A0, B, 3, 32, 32);
    }

    struct CC { int Ci, H, W, Co, pool; };
    const CC cc[8] = {{3, 32, 32, 64, 1},  {64, 16, 16, 128, 1},
                      {128, 8, 8, 256, 0}, {256, 8, 8, 256, 1},
                      {256, 4, 4, 512, 0}, {512, 4, 4, 512, 1},
                      {512, 2, 2, 512, 0}, {512, 2, 2, 512, 1}};

    char* cur = A0;
    char* nxt = A1;

    for (int i = 0; i < 8; ++i) {
        const int Ci = cc[i].Ci, H = cc[i].H, W = cc[i].W, Co = cc[i].Co;
        const float* cw = (const float*)d_in[1 + 2 * i];
        const float* cb = (const float*)d_in[2 + 2 * i];
        const int K = 9 * Ci;
        const int M = B * H * W;

        zero_u32_kernel<<<1, 1, 0, stream>>>(bits);
        absmax_kernel<<<512, TPB, 0, stream>>>(cw, Co * K, bits);
        finalize_scale_kernel<<<1, 1, 0, stream>>>(bits, 7.0f, WS);
        quant_conv_w_kernel<<<cdiv(Co * K, TPB), TPB, 0, stream>>>(
            cw, WS, (signed char*)WQ, Co, Ci);
        bias_quant_kernel<<<cdiv(Co, TPB), TPB, 0, stream>>>(cb, XS, WS, BF, Co);

        dim3 g(M / 128, Co / 64);
        if (i == 0)
            conv_iu8_kernel<true><<<g, TPB, 0, stream>>>(
                (const signed char*)cur, (const signed char*)WQ, BF, XS, WS,
                (float*)Y, H, W, Ci, Co);
        else
            conv_iu8_kernel<false><<<g, TPB, 0, stream>>>(
                (const signed char*)cur, (const signed char*)WQ, BF, XS, WS,
                (float*)Y, H, W, Ci, Co);

        int n = M * Co;
        zero_u32_kernel<<<1, 1, 0, stream>>>(bits);
        relumax_kernel<<<512, TPB, 0, stream>>>((const float*)Y, n, bits);
        finalize_scale_kernel<<<1, 1, 0, stream>>>(bits, 15.0f, XS);
        relu_quant_kernel<<<cdiv(n, TPB), TPB, 0, stream>>>(
            (const float*)Y, XS, (signed char*)nxt, n);

        if (cc[i].pool) {
            int no = B * (H / 2) * (W / 2) * Co;
            maxpool_kernel<<<cdiv(no, TPB), TPB, 0, stream>>>(
                (const signed char*)nxt, (signed char*)cur, B, H, W, Co);
        } else {
            char* t = cur; cur = nxt; nxt = t;
        }
    }

    const int LN[3] = {4096, 4096, 10};
    const int LK[3] = {512, 4096, 4096};
    for (int i = 0; i < 3; ++i) {
        const float* lw = (const float*)d_in[17 + 2 * i];
        const float* lb = (const float*)d_in[18 + 2 * i];
        const int N = LN[i], K = LK[i], M = B;

        zero_u32_kernel<<<1, 1, 0, stream>>>(bits);
        absmax_kernel<<<512, TPB, 0, stream>>>(lw, N * K, bits);
        finalize_scale_kernel<<<1, 1, 0, stream>>>(bits, 7.0f, WS);
        quant_lin_w_kernel<<<cdiv(N * K, TPB), TPB, 0, stream>>>(
            lw, WS, (signed char*)WQ, N * K);
        bias_quant_kernel<<<cdiv(N, TPB), TPB, 0, stream>>>(lb, XS, WS, BF, N);

        dim3 g(M / 128, cdiv(N, 64));
        gemm_iu8_kernel<<<g, TPB, 0, stream>>>(
            (const signed char*)cur, (const signed char*)WQ, BF, XS, WS,
            (float*)Y, M, N, K);

        if (i < 2) {
            int n = M * N;
            zero_u32_kernel<<<1, 1, 0, stream>>>(bits);
            relumax_kernel<<<512, TPB, 0, stream>>>((const float*)Y, n, bits);
            finalize_scale_kernel<<<1, 1, 0, stream>>>(bits, 15.0f, XS);
            relu_quant_kernel<<<cdiv(n, TPB), TPB, 0, stream>>>(
                (const float*)Y, XS, (signed char*)nxt, n);
            char* t = cur; cur = nxt; nxt = t;
        }
    }

    log_softmax10_kernel<<<1, TPB, 0, stream>>>((const float*)Y, (float*)d_out);
}
